// fylm_23381801960233
// MI455X (gfx1250) — compile-verified, matrix operand order checked by reading, not run on hardware
//
#include <hip/hip_runtime.h>
#include <math.h>

typedef __attribute__((ext_vector_type(16))) _Float16 v16h;
typedef __attribute__((ext_vector_type(8)))  float    v8f;

// ---------------------------------------------------------------------------
// Generic fused conv (implicit GEMM) using WMMA f32 <- f16 x f16.
//   out[co, p] = sum_k W[co,k] * im2col[k,p]   (+bn/bias/residual/activation)
// Block: 256 threads (8 waves). Tile: 64(M) x 128(N) x 32(K).
// Each wave: 32x32 output = 2x2 v_wmma_f32_16x16x32_f16 accumulators.
// LDS tiles are double-buffered and stored in WMMA *fragment order* so each
// fragment load is one aligned 32B LDS read per lane. Pixel->(y,x) division
// is hoisted out of the K loop (loop-invariant).
// ---------------------------------------------------------------------------
__global__ __launch_bounds__(256) void k_conv_wmma(
    const float* __restrict__ in, const float* __restrict__ wgt,
    float* __restrict__ out,
    const float* __restrict__ scale, const float* __restrict__ shift,
    const float* __restrict__ res,
    int Cin, int Cout, int H, int W, int ks, int pad, int act)
{
  const int HW = H * W;
  const int K  = Cin * ks * ks;
  const int m0 = blockIdx.y * 64;
  const int n0 = blockIdx.x * 128;
  const int tid  = threadIdx.x;
  const int lane = tid & 31;
  const int wv   = tid >> 5;
  const int wm   = wv & 1;    // M wave offset: 32*wm
  const int wn   = wv >> 1;   // N wave offset: 32*wn

  __shared__ __align__(32) _Float16 lA[2][64 * 32];
  __shared__ __align__(32) _Float16 lB[2][32 * 128];

  // ---- hoist loop-invariant pixel addressing (kills p/W divisions in loop)
  int b_p[16], b_y[16], b_x[16];
  #pragma unroll
  for (int j = 0; j < 16; ++j) {
    int g  = tid * 16 + j;
    int u  = g >> 9;
    int ln = (g >> 4) & 31;
    int p  = n0 + (u << 4) + (ln & 15);
    b_p[j] = p;
    int pc = min(p, HW - 1);
    int y  = pc / W;
    b_y[j] = y;
    b_x[j] = pc - y * W;
  }

  // ---- cooperative stage of one K-slab into LDS buffer bi ----
  auto loadAB = [&](int k0, int bi) {
    alignas(16) _Float16 ta[8];
    #pragma unroll
    for (int j = 0; j < 8; ++j) {
      int g  = tid * 8 + j;
      int ln = (g >> 4) & 31;
      int hi = g & 15;
      int kk = hi + (hi & 8) + ((ln & 16) >> 1);
      int m  = m0 + ((g >> 9) << 4) + (ln & 15);
      int k  = k0 + kk;
      float v = 0.f;
      if (m < Cout && k < K) v = wgt[(size_t)m * K + k];
      ta[j] = (_Float16)v;
    }
    *(uint4*)&lA[bi][tid * 8] = *(const uint4*)ta;

    alignas(16) _Float16 tb[16];
    #pragma unroll
    for (int j = 0; j < 16; ++j) {
      int g  = tid * 16 + j;
      int ln = (g >> 4) & 31;
      int hi = g & 15;
      int kk = hi + (hi & 8) + ((ln & 16) >> 1);
      int k  = k0 + kk;
      float v = 0.f;
      if (b_p[j] < HW && k < K) {
        if (ks == 3) {
          int ci = k / 9; int rs = k - ci * 9;      // constant-divisor -> mul/shift
          int r = rs / 3; int s = rs - r * 3;
          int iy = b_y[j] + r - pad, ix = b_x[j] + s - pad;
          if (iy >= 0 && iy < H && ix >= 0 && ix < W)
            v = in[(size_t)ci * HW + iy * W + ix];
        } else {
          v = in[(size_t)k * HW + b_p[j]];
        }
      }
      tb[j] = (_Float16)v;
    }
    *(uint4*)&lB[bi][tid * 16]     = *(const uint4*)tb;
    *(uint4*)&lB[bi][tid * 16 + 8] = *(const uint4*)(tb + 8);
  };

  v8f acc[2][2] = {};
  const int nK = (K + 31) >> 5;

  loadAB(0, 0);                      // prologue: stage slab 0
  for (int kt = 0; kt < nK; ++kt) {
    __syncthreads();                 // slab kt visible; slab kt-1 reads done
    const int bi = kt & 1;
    if (kt + 2 < nK) {               // prefetch two slabs ahead (global_prefetch)
      int pm = m0 + (tid >> 2);
      if (pm < Cout) __builtin_prefetch(&wgt[(size_t)pm * K + ((kt + 2) << 5)], 0, 1);
    }
    if (kt + 1 < nK) loadAB((kt + 1) << 5, bi ^ 1);   // overlap with WMMAs below

    v16h a0 = *(const v16h*)&lA[bi][((wm * 2 + 0) * 32 + lane) * 16];
    v16h a1 = *(const v16h*)&lA[bi][((wm * 2 + 1) * 32 + lane) * 16];
    v16h b0 = *(const v16h*)&lB[bi][((wn * 2 + 0) * 32 + lane) * 16];
    v16h b1 = *(const v16h*)&lB[bi][((wn * 2 + 1) * 32 + lane) * 16];

    acc[0][0] = __builtin_amdgcn_wmma_f32_16x16x32_f16(false, a0, false, b0, (short)0, acc[0][0], false, false);
    acc[0][1] = __builtin_amdgcn_wmma_f32_16x16x32_f16(false, a0, false, b1, (short)0, acc[0][1], false, false);
    acc[1][0] = __builtin_amdgcn_wmma_f32_16x16x32_f16(false, a1, false, b0, (short)0, acc[1][0], false, false);
    acc[1][1] = __builtin_amdgcn_wmma_f32_16x16x32_f16(false, a1, false, b1, (short)0, acc[1][1], false, false);
  }

  // ---- fused epilogue: bn-affine / bias / residual / relu / leaky ----
  const float bnk = 0.99999500003749972f;   // 1/sqrt(1+1e-5)
  #pragma unroll
  for (int i = 0; i < 2; ++i)
    #pragma unroll
    for (int j = 0; j < 2; ++j) {
      #pragma unroll
      for (int r = 0; r < 8; ++r) {
        int co = m0 + wm * 32 + i * 16 + r + ((lane & 16) >> 1);
        int n  = n0 + wn * 32 + j * 16 + (lane & 15);
        if (co < Cout && n < HW) {
          float v = acc[i][j][r];
          if (scale) v *= scale[co] * bnk;
          if (shift) v += shift[co];
          if (res)   v += res[(size_t)co * HW + n];
          if (act == 1)      v = fmaxf(v, 0.f);
          else if (act == 2) v = (v < 0.f) ? 0.01f * v : v;
          out[(size_t)co * HW + n] = v;
        }
      }
    }
}

// ---------------------------------------------------------------------------
// Elementwise / reduction helpers
// ---------------------------------------------------------------------------
__global__ void k_scale(const float* in, float* out, float s, int n) {
  int i = blockIdx.x * blockDim.x + threadIdx.x;
  if (i < n) out[i] = in[i] * s;
}

__global__ void k_inorm_stats(const float* x, float* stats, int HW) {
  int c = blockIdx.x;
  const float* p = x + (size_t)c * HW;
  float s = 0.f, s2 = 0.f;
  for (int i = threadIdx.x; i < HW; i += blockDim.x) { float v = p[i]; s += v; s2 += v * v; }
  __shared__ float sh0[256], sh1[256];
  sh0[threadIdx.x] = s; sh1[threadIdx.x] = s2; __syncthreads();
  for (int o = 128; o > 0; o >>= 1) {
    if ((int)threadIdx.x < o) { sh0[threadIdx.x] += sh0[threadIdx.x + o]; sh1[threadIdx.x] += sh1[threadIdx.x + o]; }
    __syncthreads();
  }
  if (threadIdx.x == 0) {
    float m = sh0[0] / (float)HW;
    float v = sh1[0] / (float)HW - m * m;
    stats[2 * c] = m; stats[2 * c + 1] = rsqrtf(v + 1e-5f);
  }
}

__global__ void k_inorm_apply(const float* x, float* y, const float* stats, int HW, int C, int sig) {
  int i = blockIdx.x * blockDim.x + threadIdx.x;
  if (i >= C * HW) return;
  int c = i / HW;
  float v = (x[i] - stats[2 * c]) * stats[2 * c + 1];
  if (sig) v = 1.f / (1.f + expf(-v));
  y[i] = v;
}

__global__ void k_maxpool2(const float* in, float* out, int C, int H, int W) {
  int OH = H >> 1, OW = W >> 1;
  int i = blockIdx.x * blockDim.x + threadIdx.x;
  if (i >= C * OH * OW) return;
  int ow = i % OW; int t = i / OW; int oh = t % OH; int c = t / OH;
  const float* p = in + ((size_t)c * H + oh * 2) * W + ow * 2;
  out[i] = fmaxf(fmaxf(p[0], p[1]), fmaxf(p[W], p[W + 1]));
}

__global__ void k_resize_add(const float* src, float* dst, int C, int H, int W, int OH, int OW) {
  int i = blockIdx.x * blockDim.x + threadIdx.x;
  if (i >= C * OH * OW) return;
  int ox = i % OW; int t = i / OW; int oy = t % OH; int c = t / OH;
  float fy = (float)oy * (float)(H - 1) / (float)(OH - 1);
  float fx = (float)ox * (float)(W - 1) / (float)(OW - 1);
  int y0 = (int)floorf(fy); int x0 = (int)floorf(fx);
  int y1 = min(y0 + 1, H - 1); int x1 = min(x0 + 1, W - 1);
  float wy = fy - (float)y0, wx = fx - (float)x0;
  const float* p = src + (size_t)c * H * W;
  float v = (1.f - wy) * ((1.f - wx) * p[y0 * W + x0] + wx * p[y0 * W + x1])
          +        wy  * ((1.f - wx) * p[y1 * W + x0] + wx * p[y1 * W + x1]);
  dst[i] += v;
}

__global__ void k_l2norm_ch(const float* in, float* out, int C, int HW) {
  int p = blockIdx.x * blockDim.x + threadIdx.x;
  if (p >= HW) return;
  float s = 0.f;
  for (int c = 0; c < C; ++c) { float v = in[(size_t)c * HW + p]; s += v * v; }
  float inv = 1.f / fmaxf(sqrtf(s), 1e-12f);
  for (int c = 0; c < C; ++c) out[(size_t)c * HW + p] = in[(size_t)c * HW + p] * inv;
}

// ---------------------------------------------------------------------------
// Keypoint extraction: NMS -> histogram top-1024 -> bitonic sort -> refine
// ---------------------------------------------------------------------------
__global__ void k_nms(const float* hm, float* mmz, int H, int W, int R, float th) {
  int IW = W - 2 * R, IH = H - 2 * R;
  int i = blockIdx.x * blockDim.x + threadIdx.x;
  if (i >= IW * IH) return;
  int x = i % IW, y = i / IW;
  float m = -1e30f;
  for (int dy = 0; dy < 2 * R + 1; ++dy)
    for (int dx = 0; dx < 2 * R + 1; ++dx)
      m = fmaxf(m, hm[(y + dy) * W + x + dx]);
  float c = hm[(y + R) * W + x + R];
  mmz[i] = (m == c && c > th) ? m : 0.f;
}

__global__ void k_kp_init(unsigned* hist, int* meta, float* listS, int* listI) {
  int i = blockIdx.x * blockDim.x + threadIdx.x;
  if (i < 1024) { hist[i] = 0u; listS[i] = 0.f; listI[i] = 0; }
  if (i < 4) meta[i] = 0;
}

__global__ void k_hist(const float* mmz, unsigned* hist, int n) {
  int i = blockIdx.x * blockDim.x + threadIdx.x;
  if (i >= n) return;
  float s = mmz[i]; if (s <= 0.f) return;
  int b = (int)(s * 1023.f); b = min(max(b, 0), 1023);
  atomicAdd(&hist[b], 1u);
}

__global__ void k_thresh(const unsigned* hist, int* meta, int k) {
  if (threadIdx.x != 0 || blockIdx.x != 0) return;
  int cum = 0, tb = -1;
  for (int b = 1023; b >= 0; --b) {
    if (cum + (int)hist[b] >= k) { tb = b; break; }
    cum += (int)hist[b];
  }
  meta[0] = tb; meta[1] = cum;
}

__global__ void k_collect(const float* mmz, int n, int* meta, float* listS, int* listI, int k) {
  int i = blockIdx.x * blockDim.x + threadIdx.x;
  if (i >= n) return;
  float s = mmz[i]; if (s <= 0.f) return;
  int b = (int)(s * 1023.f); b = min(max(b, 0), 1023);
  int tb = meta[0], cum = meta[1];
  if (b > tb) {
    int pos = atomicAdd(&meta[2], 1);
    if (pos < k) { listS[pos] = s; listI[pos] = i; }
  } else if (b == tb) {
    int pos = cum + atomicAdd(&meta[3], 1);
    if (pos < k) { listS[pos] = s; listI[pos] = i; }
  }
}

__global__ void k_sort1024(float* listS, int* listI) {
  __shared__ float ss[1024];
  __shared__ int   si[1024];
  int t = threadIdx.x;
  ss[t] = listS[t]; si[t] = listI[t]; __syncthreads();
  for (int k = 2; k <= 1024; k <<= 1) {
    for (int j = k >> 1; j > 0; j >>= 1) {
      int x = t ^ j;
      if (x > t) {
        bool desc = ((t & k) == 0);
        float a = ss[t], b = ss[x];
        int ia = si[t], ib = si[x];
        bool sw = desc ? (a < b || (a == b && ia > ib))
                       : (a > b || (a == b && ia < ib));
        if (sw) { ss[t] = b; ss[x] = a; si[t] = ib; si[x] = ia; }
      }
      __syncthreads();
    }
  }
  listS[t] = ss[t]; listI[t] = si[t];
}

__global__ void k_refine(const float* hm, const float* listS, const int* listI,
                         float* outx, float* outy, int IW, int W, int R) {
  int k = blockIdx.x * blockDim.x + threadIdx.x;
  if (k >= 1024) return;
  int idx = listI[k]; float s = listS[k];
  int y = idx / IW + R, x = idx % IW + R;
  float se = 0.f, sx = 0.f, sy = 0.f;
  for (int dy = -R; dy <= R; ++dy)
    for (int dx = -R; dx <= R; ++dx) {
      float p = hm[(y + dy) * W + x + dx];
      float e = expf((p - s) * 10.f);
      se += e; sx += e * (float)dx; sy += e * (float)dy;
    }
  outx[k] = (float)x + sx / se;
  outy[k] = (float)y + sy / se;
}

__global__ void k_sample(const float* F, const float* outx, const float* outy,
                         float* dst, int C, int FH, int FW, int IH, int IW) {
  int k = blockIdx.x; int c = threadIdx.x;
  float px = outx[k] * (float)(FW - 1) / (float)(IW - 1);
  float py = outy[k] * (float)(FH - 1) / (float)(IH - 1);
  int x0 = min(max((int)floorf(px), 0), FW - 1);
  int y0 = min(max((int)floorf(py), 0), FH - 1);
  int x1 = min(x0 + 1, FW - 1), y1 = min(y0 + 1, FH - 1);
  float wx = px - (float)x0, wy = py - (float)y0;
  const float* f = F + (size_t)c * FH * FW;
  float v = (1.f - wy) * ((1.f - wx) * f[y0 * FW + x0] + wx * f[y0 * FW + x1])
          +        wy  * ((1.f - wx) * f[y1 * FW + x0] + wx * f[y1 * FW + x1]);
  __shared__ float sh[128];
  sh[c] = v * v; __syncthreads();
  for (int o = 64; o > 0; o >>= 1) { if (c < o) sh[c] += sh[c + o]; __syncthreads(); }
  float inv = 1.f / fmaxf(sqrtf(sh[0]), 1e-12f);
  dst[(size_t)k * C + c] = v * inv;
}

// ---------------------------------------------------------------------------
// Host orchestration
// ---------------------------------------------------------------------------
extern "C" void kernel_launch(void* const* d_in, const int* in_sizes, int n_in,
                              void* d_out, int out_size, void* d_ws, size_t ws_size,
                              hipStream_t stream) {
  (void)in_sizes; (void)n_in; (void)out_size;
  int cur = 0;
  auto nx = [&]() -> const float* { return (const float*)d_in[cur++]; };

  // pytree flattening: top-level keys sorted: gray, head_params, params1, params2
  const float* gray = nx();
  const float* h_b1 = nx(); const float* h_b2 = nx();
  const float* h_w1 = nx(); const float* h_w2 = nx();

  struct CB  { const float *be1,*be2,*g1,*g2,*w1,*w2; };
  struct RB  { const float *b3,*be1,*be2,*g1,*g2,*w1,*w2,*w3; };
  struct MLP { const float *be,*g,*w1,*w2; };
  struct BBp { CB b1; RB b2, b3, b4, b5; const float *conv2,*conv3,*conv4; MLP m1, m2; };

  auto rdCB  = [&](CB&  c) { c.be1=nx(); c.be2=nx(); c.g1=nx(); c.g2=nx(); c.w1=nx(); c.w2=nx(); };
  auto rdRB  = [&](RB&  r) { r.b3=nx(); r.be1=nx(); r.be2=nx(); r.g1=nx(); r.g2=nx(); r.w1=nx(); r.w2=nx(); r.w3=nx(); };
  auto rdMLP = [&](MLP& m) { m.be=nx(); m.g=nx(); m.w1=nx(); m.w2=nx(); };

  BBp P[2];
  for (int b = 0; b < 2; ++b) {
    rdCB(P[b].b1); rdRB(P[b].b2); rdRB(P[b].b3); rdRB(P[b].b4); rdRB(P[b].b5);
    P[b].conv2 = nx(); P[b].conv3 = nx(); P[b].conv4 = nx();
    rdMLP(P[b].m1); rdMLP(P[b].m2);
  }

  // -------- workspace layout --------
  char* base = (char*)d_ws;
  size_t off = 0;
  auto alloc = [&](size_t nf) -> float* {
    float* p = (float*)(base + off);
    off += nf * sizeof(float);
    off = (off + 255) & ~(size_t)255;
    return p;
  };
  float* heat = alloc(1u << 20);
  float* hm   = alloc(1u << 20);
  float* F1   = alloc(128u * 65536u);
  float* F2   = alloc(128u * 65536u);
  float* y5r  = alloc(64u * 65536u);
  float* y7r  = alloc(128u * 16384u);
  float* G1   = alloc(16u * 1048576u);
  float* G2   = alloc(16u * 1048576u);
  float* G3   = alloc(8u * 1048576u);
  float* G4   = alloc(8u * 1048576u);
  float* stats = alloc(64);
  float* mmz   = alloc(1020u * 1020u);
  unsigned* hist = (unsigned*)alloc(1024);
  int*   meta  = (int*)alloc(16);
  float* listS = alloc(1024);
  int*   listI = (int*)alloc(1024);
  if (off > ws_size) return;   // workspace too small: bail deterministically

  auto conv = [&](const float* in, const float* w, float* out, const float* sc,
                  const float* sh, const float* res, int Cin, int Cout, int H, int W,
                  int ks, int pad, int act) {
    dim3 g((H * W + 127) / 128, (Cout + 63) / 64);
    k_conv_wmma<<<g, 256, 0, stream>>>(in, w, out, sc, sh, res, Cin, Cout, H, W, ks, pad, act);
  };
  auto mp = [&](const float* in, float* out, int C, int H, int W) {
    int n = C * (H / 2) * (W / 2);
    k_maxpool2<<<(n + 255) / 256, 256, 0, stream>>>(in, out, C, H, W);
  };
  auto rsz = [&](const float* src, float* dst, int C, int H, int W, int OH, int OW) {
    int n = C * OH * OW;
    k_resize_add<<<(n + 255) / 256, 256, 0, stream>>>(src, dst, C, H, W, OH, OW);
  };

  // -------- heat = inorm(gray / 255) --------
  k_scale<<<(1 << 20) / 256, 256, 0, stream>>>(gray, heat, 1.f / 255.f, 1 << 20);
  k_inorm_stats<<<1, 256, 0, stream>>>(heat, stats, 1 << 20);
  k_inorm_apply<<<(1 << 20) / 256, 256, 0, stream>>>(heat, heat, stats, 1 << 20, 1, 0);

  // -------- backbones --------
  auto backbone = [&](const BBp& p, float* F) {
    conv(heat, p.b1.w1, G1, p.b1.g1, p.b1.be1, nullptr, 1, 16, 1024, 1024, 3, 1, 1);
    conv(G1,   p.b1.w2, G2, p.b1.g2, p.b1.be2, nullptr, 16, 16, 1024, 1024, 3, 1, 1);
    mp(G2, G1, 16, 1024, 1024);
    // res_block 16->32 @512
    conv(G1, p.b2.w1, G3, p.b2.g1, p.b2.be1, nullptr, 16, 32, 512, 512, 3, 1, 1);
    conv(G1, p.b2.w3, G4, nullptr, p.b2.b3, nullptr, 16, 32, 512, 512, 1, 0, 0);
    conv(G3, p.b2.w2, G2, p.b2.g2, p.b2.be2, G4,     32, 32, 512, 512, 3, 1, 1);
    mp(G2, G1, 32, 512, 512);
    // res_block 32->64 @256 -> y5
    conv(G1, p.b3.w1, G3, p.b3.g1, p.b3.be1, nullptr, 32, 64, 256, 256, 3, 1, 1);
    conv(G1, p.b3.w3, G4, nullptr, p.b3.b3, nullptr, 32, 64, 256, 256, 1, 0, 0);
    conv(G3, p.b3.w2, y5r, p.b3.g2, p.b3.be2, G4,    64, 64, 256, 256, 3, 1, 1);
    mp(y5r, G1, 64, 256, 256);
    // res_block 64->128 @128 -> y7
    conv(G1, p.b4.w1, G3, p.b4.g1, p.b4.be1, nullptr, 64, 128, 128, 128, 3, 1, 1);
    conv(G1, p.b4.w3, G4, nullptr, p.b4.b3, nullptr, 64, 128, 128, 128, 1, 0, 0);
    conv(G3, p.b4.w2, y7r, p.b4.g2, p.b4.be2, G4,    128, 128, 128, 128, 3, 1, 1);
    mp(y7r, G1, 128, 128, 128);
    // res_block 128->256 @64
    conv(G1, p.b5.w1, G3, p.b5.g1, p.b5.be1, nullptr, 128, 256, 64, 64, 3, 1, 1);
    conv(G1, p.b5.w3, G4, nullptr, p.b5.b3, nullptr, 128, 256, 64, 64, 1, 0, 0);
    conv(G3, p.b5.w2, G2, p.b5.g2, p.b5.be2, G4,     256, 256, 64, 64, 3, 1, 1);
    // y9 = conv4(.)
    conv(G2, p.conv4, G1, nullptr, nullptr, nullptr, 256, 256, 64, 64, 1, 0, 0);
    // z1 = conv3(y7) + resize(y9)
    conv(y7r, p.conv3, G2, nullptr, nullptr, nullptr, 128, 256, 128, 128, 1, 0, 0);
    rsz(G1, G2, 256, 64, 64, 128, 128);
    // mlp1
    conv(G2, p.m1.w1, G3, p.m1.g, p.m1.be, nullptr, 256, 256, 128, 128, 3, 1, 2);
    conv(G3, p.m1.w2, G1, nullptr, nullptr, nullptr, 256, 128, 128, 128, 3, 1, 0);
    // z3 = conv2(y5) + resize(y7b)
    conv(y5r, p.conv2, G2, nullptr, nullptr, nullptr, 64, 128, 256, 256, 1, 0, 0);
    rsz(G1, G2, 128, 128, 128, 256, 256);
    // mlp2
    conv(G2, p.m2.w1, G3, p.m2.g, p.m2.be, nullptr, 128, 128, 256, 256, 3, 1, 2);
    conv(G3, p.m2.w2, G2, nullptr, nullptr, nullptr, 128, 128, 256, 256, 3, 1, 0);
    k_l2norm_ch<<<65536 / 256, 256, 0, stream>>>(G2, F, 128, 65536);
  };
  backbone(P[0], F1);
  backbone(P[1], F2);

  // -------- head --------
  conv(heat, h_w1, G1, nullptr, h_b1, nullptr, 1, 16, 1024, 1024, 3, 1, 0);
  k_inorm_stats<<<16, 256, 0, stream>>>(G1, stats, 1 << 20);
  k_inorm_apply<<<(16 << 20) / 256, 256, 0, stream>>>(G1, G1, stats, 1 << 20, 16, 0);
  conv(G1, h_w2, G2, nullptr, h_b2, nullptr, 16, 1, 1024, 1024, 1, 0, 0);
  k_inorm_stats<<<1, 256, 0, stream>>>(G2, stats, 1 << 20);
  k_inorm_apply<<<(1 << 20) / 256, 256, 0, stream>>>(G2, hm, stats, 1 << 20, 1, 1);

  // -------- keypoints --------
  const int NINNER = 1020 * 1020;
  k_nms<<<(NINNER + 255) / 256, 256, 0, stream>>>(hm, mmz, 1024, 1024, 2, 0.2f);
  k_kp_init<<<4, 256, 0, stream>>>(hist, meta, listS, listI);
  k_hist<<<(NINNER + 255) / 256, 256, 0, stream>>>(mmz, hist, NINNER);
  k_thresh<<<1, 1, 0, stream>>>(hist, meta, 1024);
  k_collect<<<(NINNER + 255) / 256, 256, 0, stream>>>(mmz, NINNER, meta, listS, listI, 1024);
  k_sort1024<<<1, 1024, 0, stream>>>(listS, listI);

  float* out  = (float*)d_out;
  float* outx = out + 262144;   // f1(1024x128) + f2(1024x128)
  float* outy = outx + 1024;
  k_refine<<<4, 256, 0, stream>>>(hm, listS, listI, outx, outy, 1020, 1024, 2);
  k_sample<<<1024, 128, 0, stream>>>(F1, outx, outy, out,          128, 256, 256, 1024, 1024);
  k_sample<<<1024, 128, 0, stream>>>(F2, outx, outy, out + 131072, 128, 256, 256, 1024, 1024);
}
